// PromptEmbedding_69595650064833
// MI455X (gfx1250) — matrix operand order repaired and checked by reading, not compiled, not run on hardware
//
#include <hip/hip_runtime.h>

// ---------------------------------------------------------------------------
// Types for CDNA5 WMMA (wave32): v_wmma_f32_16x16x32_bf16
// ---------------------------------------------------------------------------
typedef __attribute__((ext_vector_type(16))) __bf16 bf16x16;
typedef __attribute__((ext_vector_type(8)))  __bf16 bf16x8;
typedef __attribute__((ext_vector_type(8)))  float  f32x8;

// A-fragment (16x32 bf16, MxK) from LDS tile stored row-major [16+][stride].
// ISA layout: lanes 0-15 row M=lane hold K0..7 (V0-3) and K16..23 (V4-7);
// lanes 16-31 row M=lane-16 hold K8..15 and K24..31.
__device__ __forceinline__ bf16x16 frag_a_lds(const __bf16* base, int stride) {
  const int lane = threadIdx.x & 31;
  const __bf16* p = base + (lane & 15) * stride + ((lane >> 4) << 3);
  bf16x8 lo = *(const bf16x8*)p;         // K kb..kb+7
  bf16x8 hi = *(const bf16x8*)(p + 16);  // K kb+16..kb+23
  return __builtin_shufflevector(lo, hi, 0,1,2,3,4,5,6,7,8,9,10,11,12,13,14,15);
}

// B-fragment (32x16 bf16, KxN) from LDS tile stored TRANSPOSED [N][stride(K)].
// ISA layout: lanes 0-15 col N=lane hold K0..7 and K16..23; lanes 16-31 hold K8..15 and K24..31.
__device__ __forceinline__ bf16x16 frag_b_lds(const __bf16* baseT, int stride) {
  const int lane = threadIdx.x & 31;
  const __bf16* p = baseT + (lane & 15) * stride + ((lane >> 4) << 3);
  bf16x8 lo = *(const bf16x8*)p;
  bf16x8 hi = *(const bf16x8*)(p + 16);
  return __builtin_shufflevector(lo, hi, 0,1,2,3,4,5,6,7,8,9,10,11,12,13,14,15);
}

__device__ __forceinline__ f32x8 wmma_bf16(bf16x16 a, bf16x16 b, f32x8 c) {
  return __builtin_amdgcn_wmma_f32_16x16x32_bf16(false, a, false, b, (short)0, c,
                                                 false, false);
}

__device__ __forceinline__ float gelu_exact(float x) {
  return 0.5f * x * (1.0f + erff(x * 0.7071067811865476f));
}

// ---------------------------------------------------------------------------
// Generic GEMM: C[M,N] = op(A[M,K] @ B[K,N] + bias[N])
// A,B fp32 row-major in global; converted to bf16 while staging into LDS.
// Block tile 64x128, K-step 32. 8 waves in a 2(M)x4(N) grid; each wave owns a
// 32x32 register tile = 2x2 WMMA tiles with A/B fragment reuse (4 WMMA per
// K-step per wave). Global loads for the next K-tile are issued between the
// barriers (software pipelining), plus a prefetch 2 tiles ahead.
// mode: 0 = identity, 1 = exact GELU
// Requires: K % 32 == 0, N % 128 == 0 (true for all uses here). M guarded.
// ---------------------------------------------------------------------------
__global__ __launch_bounds__(256) void gemm_bf16_wmma(
    const float* __restrict__ A, const float* __restrict__ Bm,
    const float* __restrict__ bias, float* __restrict__ C,
    int M, int N, int K, int mode)
{
  constexpr int BM = 64, BN = 128, BK = 32;
  constexpr int AS = BK + 8;   // LDS row stride (bf16) to rotate banks
  constexpr int BS = BK + 8;
  __shared__ __bf16 At[BM * AS];      // [m][k]
  __shared__ __bf16 Bt[BN * BS];      // [n][k]  (transposed)

  const int tid  = threadIdx.x;
  const int lane = tid & 31;
  const int w    = tid >> 5;          // 8 waves
  const int wm   = w & 1;             // 2 row groups of 32
  const int wn   = w >> 1;            // 4 col groups of 32
  const int m0   = blockIdx.y * BM;
  const int n0   = blockIdx.x * BN;

  // staging thread mapping
  const int ar = tid >> 2, ac = (tid & 3) << 3;     // A: 64 rows x 32 cols, 8 f/thr
  const int bk = tid >> 3, bn = (tid & 7) << 4;     // B: 32 rows x 128 cols, 16 f/thr
  const bool aValid = (m0 + ar) < M;
  const float* aPtr = A + (size_t)(m0 + ar) * K + ac;
  const float* bPtr = Bm + (size_t)bk * N + n0 + bn;

  f32x8 acc00 = {}, acc01 = {}, acc10 = {}, acc11 = {};

  // preload first K-tile into registers
  float4 rA0 = make_float4(0.f,0.f,0.f,0.f), rA1 = rA0;
  float4 rB0, rB1, rB2, rB3;
  if (aValid) { rA0 = *(const float4*)(aPtr); rA1 = *(const float4*)(aPtr + 4); }
  rB0 = *(const float4*)(bPtr);      rB1 = *(const float4*)(bPtr + 4);
  rB2 = *(const float4*)(bPtr + 8);  rB3 = *(const float4*)(bPtr + 12);

  for (int k0 = 0; k0 < K; k0 += BK) {
    // ---- store staged registers to LDS (bf16) ----
    {
      __bf16* d = At + ar * AS + ac;
      d[0]=(__bf16)rA0.x; d[1]=(__bf16)rA0.y; d[2]=(__bf16)rA0.z; d[3]=(__bf16)rA0.w;
      d[4]=(__bf16)rA1.x; d[5]=(__bf16)rA1.y; d[6]=(__bf16)rA1.z; d[7]=(__bf16)rA1.w;
      const float bv[16] = { rB0.x,rB0.y,rB0.z,rB0.w, rB1.x,rB1.y,rB1.z,rB1.w,
                             rB2.x,rB2.y,rB2.z,rB2.w, rB3.x,rB3.y,rB3.z,rB3.w };
      for (int j = 0; j < 16; ++j) Bt[(bn + j) * BS + bk] = (__bf16)bv[j];
    }
    __syncthreads();

    // ---- issue next K-tile global loads (overlap with WMMA) ----
    const bool more = (k0 + BK) < K;
    float4 nA0 = make_float4(0.f,0.f,0.f,0.f), nA1 = nA0;
    float4 nB0 = nA0, nB1 = nA0, nB2 = nA0, nB3 = nA0;
    if (more) {
      const float* ap = aPtr + (k0 + BK);
      if (aValid) { nA0 = *(const float4*)(ap); nA1 = *(const float4*)(ap + 4); }
      const float* bp = bPtr + (size_t)(k0 + BK) * N;
      nB0 = *(const float4*)(bp);      nB1 = *(const float4*)(bp + 4);
      nB2 = *(const float4*)(bp + 8);  nB3 = *(const float4*)(bp + 12);
      if (k0 + 2 * BK < K)
        __builtin_prefetch(bPtr + (size_t)(k0 + 2 * BK) * N, 0, 1);
    }

    // ---- 2x2 WMMA tiles with fragment reuse ----
    bf16x16 a0 = frag_a_lds(At + (wm * 32) * AS, AS);
    bf16x16 a1 = frag_a_lds(At + (wm * 32 + 16) * AS, AS);
    bf16x16 b0 = frag_b_lds(Bt + (wn * 32) * BS, BS);
    bf16x16 b1 = frag_b_lds(Bt + (wn * 32 + 16) * BS, BS);
    acc00 = wmma_bf16(a0, b0, acc00);
    acc01 = wmma_bf16(a0, b1, acc01);
    acc10 = wmma_bf16(a1, b0, acc10);
    acc11 = wmma_bf16(a1, b1, acc11);
    __syncthreads();

    rA0 = nA0; rA1 = nA1; rB0 = nB0; rB1 = nB1; rB2 = nB2; rB3 = nB3;
  }

  // ---- epilogue: D layout -> lane = N col (mod 16), VGPR v -> M = v (+8) ----
  const int cBase = n0 + wn * 32 + (lane & 15);
  const int rBase = m0 + wm * 32 + ((lane >> 4) << 3);
  for (int ti = 0; ti < 2; ++ti) {
    for (int tj = 0; tj < 2; ++tj) {
      const f32x8 acc = (ti == 0) ? (tj == 0 ? acc00 : acc01)
                                  : (tj == 0 ? acc10 : acc11);
      const int col = cBase + tj * 16;
      const float bv = bias ? bias[col] : 0.f;
      for (int v = 0; v < 8; ++v) {
        const int row = rBase + ti * 16 + v;
        if (row < M) {
          float x = acc[v] + bv;
          if (mode == 1) x = gelu_exact(x);
          C[(size_t)row * N + col] = x;
        }
      }
    }
  }
}

// ---------------------------------------------------------------------------
// Fused self-attention per (b,h): S = QK^T/32 (64x64), mask+softmax, O = P V.
// Q,K,V layout (B, LN, H*D) fp32. One block of 256 threads per (b,h).
// ---------------------------------------------------------------------------
__global__ __launch_bounds__(256) void attn_kernel(
    const float* __restrict__ Q, const float* __restrict__ Km,
    const float* __restrict__ V, const int* __restrict__ gmask,
    float* __restrict__ O)
{
  constexpr int QS = 40, SS = 66, PS = 72, VS = 72;
  __shared__ __bf16 Qt[64 * QS];   // [q][k32]
  __shared__ __bf16 Kt[64 * QS];   // [key][k32]  == B^T layout for S=QK^T
  __shared__ float  S [64 * SS];   // scores / probs scratch
  __shared__ __bf16 Pb[64 * PS];   // probs bf16 [q][key]
  __shared__ __bf16 Vt[64 * VS];   // [dcol][key] == B^T layout for O=PV

  const int b = blockIdx.x >> 2, h = blockIdx.x & 3;
  const int tid = threadIdx.x, lane = tid & 31, w = tid >> 5;
  const int rm = w & 3;            // 4 row tiles of 16
  const int cn = w >> 2;           // waves own col tiles cn and cn+2
  const size_t base = ((size_t)b * 64) * 4096 + (size_t)h * 1024;

  // ---- phase 1: S = Q K^T over D=1024 ----
  f32x8 s0 = {}, s1 = {};
  for (int k0 = 0; k0 < 1024; k0 += 32) {
    const int l = tid >> 2, c8 = (tid & 3) * 8;
    {
      const float* pq = Q + base + (size_t)l * 4096 + k0 + c8;
      float4 a0 = *(const float4*)pq, a1 = *(const float4*)(pq + 4);
      __bf16* d = Qt + l * QS + c8;
      d[0]=(__bf16)a0.x; d[1]=(__bf16)a0.y; d[2]=(__bf16)a0.z; d[3]=(__bf16)a0.w;
      d[4]=(__bf16)a1.x; d[5]=(__bf16)a1.y; d[6]=(__bf16)a1.z; d[7]=(__bf16)a1.w;
      const float* pk = Km + base + (size_t)l * 4096 + k0 + c8;
      float4 b0 = *(const float4*)pk, b1 = *(const float4*)(pk + 4);
      __bf16* e = Kt + l * QS + c8;
      e[0]=(__bf16)b0.x; e[1]=(__bf16)b0.y; e[2]=(__bf16)b0.z; e[3]=(__bf16)b0.w;
      e[4]=(__bf16)b1.x; e[5]=(__bf16)b1.y; e[6]=(__bf16)b1.z; e[7]=(__bf16)b1.w;
    }
    __syncthreads();
    bf16x16 a  = frag_a_lds(Qt + rm * 16 * QS, QS);
    bf16x16 bA = frag_b_lds(Kt + (cn * 16) * QS, QS);
    bf16x16 bB = frag_b_lds(Kt + ((cn + 2) * 16) * QS, QS);
    s0 = wmma_bf16(a, bA, s0);
    s1 = wmma_bf16(a, bB, s1);
    __syncthreads();
  }
  {
    const int cA = cn * 16 + (lane & 15), cB = (cn + 2) * 16 + (lane & 15);
    const int rb = rm * 16 + ((lane >> 4) << 3);
    for (int v = 0; v < 8; ++v) {
      S[(rb + v) * SS + cA] = s0[v] * 0.03125f;   // 1/sqrt(1024)
      S[(rb + v) * SS + cB] = s1[v] * 0.03125f;
    }
  }
  __syncthreads();

  // ---- masked softmax per row (threads 0..63) ----
  if (tid < 64) {
    float mx = -INFINITY;
    for (int k2 = 0; k2 < 64; ++k2) {
      float sv = (gmask[b * 64 + k2] == 0) ? -INFINITY : S[tid * SS + k2];
      mx = fmaxf(mx, sv);
    }
    float sum = 0.f;
    for (int k2 = 0; k2 < 64; ++k2) {
      float sv = (gmask[b * 64 + k2] == 0) ? -INFINITY : S[tid * SS + k2];
      float p = expf(sv - mx);
      sum += p;
      S[tid * SS + k2] = p;
    }
    const float inv = 1.f / sum;
    for (int k2 = 0; k2 < 64; ++k2)
      Pb[tid * PS + k2] = (__bf16)(S[tid * SS + k2] * inv);
  }
  __syncthreads();

  // ---- phase 2: O = P @ V, 64-col chunks of D ----
  for (int c0 = 0; c0 < 1024; c0 += 64) {
    {
      const int key = tid >> 2, cb = (tid & 3) * 16;
      const float* pv = V + ((size_t)(b * 64 + key)) * 4096 + h * 1024 + c0 + cb;
      for (int j = 0; j < 16; j += 4) {
        float4 v = *(const float4*)(pv + j);
        Vt[(cb + j + 0) * VS + key] = (__bf16)v.x;
        Vt[(cb + j + 1) * VS + key] = (__bf16)v.y;
        Vt[(cb + j + 2) * VS + key] = (__bf16)v.z;
        Vt[(cb + j + 3) * VS + key] = (__bf16)v.w;
      }
    }
    __syncthreads();
    f32x8 o0 = {}, o1 = {};
    for (int kk = 0; kk < 64; kk += 32) {
      bf16x16 a  = frag_a_lds(Pb + rm * 16 * PS + kk, PS);
      bf16x16 bA = frag_b_lds(Vt + (cn * 16) * VS + kk, VS);
      bf16x16 bB = frag_b_lds(Vt + ((cn + 2) * 16) * VS + kk, VS);
      o0 = wmma_bf16(a, bA, o0);
      o1 = wmma_bf16(a, bB, o1);
    }
    const int cA = c0 + cn * 16 + (lane & 15), cB = c0 + (cn + 2) * 16 + (lane & 15);
    const int rb = rm * 16 + ((lane >> 4) << 3);
    for (int v = 0; v < 8; ++v) {
      const size_t ro = ((size_t)(b * 64 + rb + v)) * 4096 + (size_t)h * 1024;
      O[ro + cA] = o0[v];
      O[ro + cB] = o1[v];
    }
    __syncthreads();
  }
}

// ---------------------------------------------------------------------------
// Residual add + LayerNorm over D=1024. One block per row.
// ---------------------------------------------------------------------------
__global__ __launch_bounds__(256) void add_ln_kernel(
    const float* __restrict__ Y, const float* __restrict__ X,
    const float* __restrict__ g, const float* __restrict__ bt,
    float* __restrict__ H2)
{
  const int row = blockIdx.x, t = threadIdx.x;
  __shared__ float red[256];
  float v[4], s = 0.f;
  for (int j = 0; j < 4; ++j) {
    const int d = t + 256 * j;
    v[j] = Y[(size_t)row * 1024 + d] + X[(size_t)row * 1024 + d];
    s += v[j];
  }
  red[t] = s; __syncthreads();
  for (int st = 128; st > 0; st >>= 1) { if (t < st) red[t] += red[t + st]; __syncthreads(); }
  const float mu = red[0] * (1.f / 1024.f);
  __syncthreads();
  float q = 0.f;
  for (int j = 0; j < 4; ++j) { const float d = v[j] - mu; q += d * d; }
  red[t] = q; __syncthreads();
  for (int st = 128; st > 0; st >>= 1) { if (t < st) red[t] += red[t + st]; __syncthreads(); }
  const float inv = rsqrtf(red[0] * (1.f / 1024.f) + 1e-5f);
  for (int j = 0; j < 4; ++j) {
    const int d = t + 256 * j;
    H2[(size_t)row * 1024 + d] = (v[j] - mu) * inv * g[d] + bt[d];
  }
}

// ---------------------------------------------------------------------------
// Additive (Bahdanau) attention: one block per query row (B*LN = 512 blocks).
// e[k] = sum_d att_v[d]*tanh(q[d]+kproj[k][d]+att_b[d]); masked softmax over
// LT=256 keys (one key per thread); H3 = w @ text_embeds. tanh-bound (TRANS).
// ---------------------------------------------------------------------------
__global__ __launch_bounds__(256) void addattn_kernel(
    const float* __restrict__ qp, const float* __restrict__ kp,
    const float* __restrict__ att_v, const float* __restrict__ att_b,
    const int* __restrict__ lmask, const float* __restrict__ text,
    float* __restrict__ H3)
{
  const int blk = blockIdx.x, b = blk >> 6, t = threadIdx.x;
  __shared__ float qs[1024], avs[1024], abv[1024], red[256], ws[256];
  for (int j = 0; j < 4; ++j) {
    const int d = t + 256 * j;
    qs[d]  = qp[(size_t)blk * 1024 + d];
    avs[d] = att_v[d];
    abv[d] = att_b[d];
  }
  __syncthreads();

  float e;
  {
    const float* krow = kp + ((size_t)(b * 256 + t)) * 1024;
    float acc = 0.f;
    for (int d = 0; d < 1024; ++d)
      acc += avs[d] * tanhf(qs[d] + krow[d] + abv[d]);
    e = (lmask[b * 256 + t] == 0) ? -INFINITY : acc;
  }
  red[t] = e; __syncthreads();
  for (int st = 128; st > 0; st >>= 1) { if (t < st) red[t] = fmaxf(red[t], red[t + st]); __syncthreads(); }
  const float mx = red[0]; __syncthreads();
  const float p = expf(e - mx);
  red[t] = p; __syncthreads();
  for (int st = 128; st > 0; st >>= 1) { if (t < st) red[t] += red[t + st]; __syncthreads(); }
  const float sum = red[0]; __syncthreads();
  ws[t] = p / sum; __syncthreads();

  for (int j = 0; j < 4; ++j) {
    const int d = t + 256 * j;
    float acc = 0.f;
    for (int k = 0; k < 256; ++k)
      acc += ws[k] * text[((size_t)b * 256 + k) * 1024 + d];
    H3[(size_t)blk * 1024 + d] = acc;
  }
}

// ---------------------------------------------------------------------------
// Masked mean pool over nodes: H4[b][d]. One block per batch b.
// ---------------------------------------------------------------------------
__global__ __launch_bounds__(256) void pool_kernel(
    const float* __restrict__ H3, const int* __restrict__ gmask,
    float* __restrict__ H4)
{
  const int b = blockIdx.x, t = threadIdx.x;
  float cnt = 0.f;
  for (int l = 0; l < 64; ++l) cnt += (gmask[b * 64 + l] != 0) ? 1.f : 0.f;
  cnt = fmaxf(cnt, 0.0009765625f);   // np.finfo(float16).eps
  const float inv = 1.f / cnt;
  for (int j = 0; j < 4; ++j) {
    const int d = t + 256 * j;
    float acc = 0.f;
    for (int l = 0; l < 64; ++l)
      if (gmask[b * 64 + l] != 0) acc += H3[((size_t)b * 64 + l) * 1024 + d];
    H4[(size_t)b * 1024 + d] = acc * inv;
  }
}

// ---------------------------------------------------------------------------
extern "C" void kernel_launch(void* const* d_in, const int* in_sizes, int n_in,
                              void* d_out, int out_size, void* d_ws, size_t ws_size,
                              hipStream_t stream)
{
  const float* text = (const float*)d_in[0];   // (8,256,1024)
  const float* node = (const float*)d_in[1];   // (8,64,1024)
  const float* Wq   = (const float*)d_in[2];
  const float* bq   = (const float*)d_in[3];
  const float* Wk   = (const float*)d_in[4];
  const float* bk   = (const float*)d_in[5];
  const float* Wv   = (const float*)d_in[6];
  const float* bv   = (const float*)d_in[7];
  const float* Wo   = (const float*)d_in[8];
  const float* bo   = (const float*)d_in[9];
  const float* ln_g = (const float*)d_in[10];
  const float* ln_b = (const float*)d_in[11];
  const float* aqW  = (const float*)d_in[12];
  const float* aqb  = (const float*)d_in[13];
  const float* akW  = (const float*)d_in[14];
  const float* akb  = (const float*)d_in[15];
  const float* attv = (const float*)d_in[16];
  const float* attb = (const float*)d_in[17];
  const float* dp1W = (const float*)d_in[18];
  const float* dp1b = (const float*)d_in[19];
  const float* dp2W = (const float*)d_in[20];
  const float* dp2b = (const float*)d_in[21];
  const int*   gmask = (const int*)d_in[22];   // (8,64)
  const int*   lmask = (const int*)d_in[23];   // (8,256)
  float* out = (float*)d_out;                  // (8,2048)

  // scratch layout (floats)
  float* ws  = (float*)d_ws;
  float* Qb  = ws;              // 512*4096
  float* Kb  = Qb  + 2097152;   // 512*4096
  float* Vb  = Kb  + 2097152;   // 512*4096
  float* AO  = Vb  + 2097152;   // 512*4096 attn out (B,LN,H*D)
  float* WoO = AO  + 2097152;   // 512*1024
  float* H2  = WoO + 524288;    // 512*1024
  float* AQ  = H2  + 524288;    // 512*1024
  float* AKP = AQ  + 524288;    // 2048*1024
  float* H3  = AKP + 2097152;   // 512*1024
  float* H4  = H3  + 524288;    // 8*1024
  float* Z1  = H4  + 8192;      // 8*1024

  const dim3 blk(256);

  // QKV projections: (512,1024) @ (1024,4096); block tile 64x128
  gemm_bf16_wmma<<<dim3(32, 8), blk, 0, stream>>>(node, Wq, bq, Qb, 512, 4096, 1024, 0);
  gemm_bf16_wmma<<<dim3(32, 8), blk, 0, stream>>>(node, Wk, bk, Kb, 512, 4096, 1024, 0);
  gemm_bf16_wmma<<<dim3(32, 8), blk, 0, stream>>>(node, Wv, bv, Vb, 512, 4096, 1024, 0);

  // fused attention per (b,h)
  attn_kernel<<<dim3(32), blk, 0, stream>>>(Qb, Kb, Vb, gmask, AO);

  // output projection: (512,4096) @ (4096,1024)
  gemm_bf16_wmma<<<dim3(8, 8), blk, 0, stream>>>(AO, Wo, bo, WoO, 512, 1024, 4096, 0);

  // residual + LayerNorm
  add_ln_kernel<<<dim3(512), blk, 0, stream>>>(WoO, node, ln_g, ln_b, H2);

  // additive attention projections
  gemm_bf16_wmma<<<dim3(8, 8), blk, 0, stream>>>(H2, aqW, aqb, AQ, 512, 1024, 1024, 0);
  gemm_bf16_wmma<<<dim3(8, 32), blk, 0, stream>>>(text, akW, akb, AKP, 2048, 1024, 1024, 0);

  // tanh additive attention + weighted sum of text embeds
  addattn_kernel<<<dim3(512), blk, 0, stream>>>(AQ, AKP, attv, attb, lmask, text, H3);

  // masked mean pool
  pool_kernel<<<dim3(8), blk, 0, stream>>>(H3, gmask, H4);

  // domain projector: GELU GEMM then final GEMM into d_out
  gemm_bf16_wmma<<<dim3(8, 1), blk, 0, stream>>>(H4, dp1W, dp1b, Z1, 8, 1024, 1024, 1);
  gemm_bf16_wmma<<<dim3(16, 1), blk, 0, stream>>>(Z1, dp2W, dp2b, out, 8, 2048, 1024, 0);
}
